// StatefulDPDFNet_48k_29557964931367
// MI455X (gfx1250) — matrix operand order repaired and checked by reading, not compiled, not run on hardware
//
// StatefulDPDFNet (DeepFilterNet-style) single-step inference for MI455X (gfx1250).
//
// Roofline: ~45M MACs, ~180MB fp32 weight traffic per call -> pure bandwidth bound
// (~7.7us @ 23.3TB/s HBM; model fits the 192MB L2 so replays stream from L2).
// All big matvecs run through a wave32 WMMA GEMV. Preferred path: native fp32
// V_WMMA_F32_16X16X4_F32 (probed via __has_builtin); fallback: codegen-confirmed
// V_WMMA_F32_16X16X32_F16 with in-register f32->f16 convert. Dual accumulators
// break the WMMA->WMMA dependency chain; global_prefetch_b8 streams ahead.
#include <hip/hip_runtime.h>
#include <math.h>

typedef __attribute__((ext_vector_type(16))) _Float16 v16h;
typedef __attribute__((ext_vector_type(8)))  float    v8f;
typedef __attribute__((ext_vector_type(2)))  float    v2f;

#define ACT_NONE 0
#define ACT_RELU 1
#define ACT_SIG  2

__device__ __forceinline__ float sigmf(float x){ return 1.f/(1.f+expf(-x)); }

#if __has_builtin(__builtin_amdgcn_wmma_f32_16x16x4_f32)
#define GEMV_F32X4 1
// y[0..O) = act(W[OxI] @ x + bias (+ addv)) * scale + off.  One wave per block,
// 16 rows per wave, native fp32 WMMA, K consumed 4 per instruction.
// 32-bit A 16x4 lane layout (ISA 7.12.2): lane<16 holds K{0,1}, lane>=16 holds
// K{2,3} for row M = lane&15; B (4x16) mirrors with x replicated over N.
__global__ void k_gemv_wmma(const float* __restrict__ W, const float* __restrict__ x,
                            const float* __restrict__ bias, const float* __restrict__ addv,
                            float* __restrict__ y, int O, int I, int act,
                            float scale, float off)
{
  const int lane = threadIdx.x;
  const int m = lane & 15, hi = lane >> 4;
  int row = blockIdx.x * 16 + m; if (row >= O) row = O - 1;   // clamp; guarded store
  const float* Wr = W + (size_t)row * I;
  const int kk = 2*hi;                       // this lane's K pair within a 4-chunk
  v8f acc0 = {}, acc1 = {};
  for (int k0 = 0; k0 < I; k0 += 32) {       // I % 32 == 0 for every layer here
    __builtin_prefetch(Wr + k0 + 2048, 0, 0);                // global_prefetch_b8
    #pragma unroll
    for (int kc = 0; kc < 32; kc += 8) {
      v2f a0 = *reinterpret_cast<const v2f*>(Wr + k0 + kc + kk);
      v2f b0 = *reinterpret_cast<const v2f*>(x  + k0 + kc + kk);
      acc0 = __builtin_amdgcn_wmma_f32_16x16x4_f32(false, a0, false, b0, (short)0,
                                                   acc0, false, false);
      v2f a1 = *reinterpret_cast<const v2f*>(Wr + k0 + kc + 4 + kk);
      v2f b1 = *reinterpret_cast<const v2f*>(x  + k0 + kc + 4 + kk);
      acc1 = __builtin_amdgcn_wmma_f32_16x16x4_f32(false, a1, false, b1, (short)0,
                                                   acc1, false, false);
    }
  }
  v8f acc = acc0 + acc1;
  // D layout: lanes 0/16 hold y rows (M = hi*8 + r) in acc[r]; all N columns equal.
  if (m == 0) {
    #pragma unroll
    for (int r = 0; r < 8; ++r) {
      int orow = blockIdx.x*16 + hi*8 + r;
      if (orow < O) {
        float v = acc[r] + bias[orow];
        if (addv) v += addv[orow];
        if (act == ACT_RELU) v = fmaxf(v, 0.f);
        else if (act == ACT_SIG) v = sigmf(v);
        y[orow] = v * scale + off;
      }
    }
  }
}
#else
__device__ __forceinline__ v16h cvt16(float4 a0, float4 a1, float4 a2, float4 a3){
  v16h r;
  r[0]=(_Float16)a0.x;  r[1]=(_Float16)a0.y;  r[2]=(_Float16)a0.z;  r[3]=(_Float16)a0.w;
  r[4]=(_Float16)a1.x;  r[5]=(_Float16)a1.y;  r[6]=(_Float16)a1.z;  r[7]=(_Float16)a1.w;
  r[8]=(_Float16)a2.x;  r[9]=(_Float16)a2.y;  r[10]=(_Float16)a2.z; r[11]=(_Float16)a2.w;
  r[12]=(_Float16)a3.x; r[13]=(_Float16)a3.y; r[14]=(_Float16)a3.z; r[15]=(_Float16)a3.w;
  return r;
}
// Fallback: f16 WMMA (f32 accumulate), dual accumulators over 64-wide K steps.
// 16-bit A 16x32 lane layout: lane<16 -> K {0..7,16..23}, lane>=16 -> K {8..15,24..31}.
__global__ void k_gemv_wmma(const float* __restrict__ W, const float* __restrict__ x,
                            const float* __restrict__ bias, const float* __restrict__ addv,
                            float* __restrict__ y, int O, int I, int act,
                            float scale, float off)
{
  const int lane = threadIdx.x;
  const int m = lane & 15, hi = lane >> 4;
  int row = blockIdx.x * 16 + m; if (row >= O) row = O - 1;
  const float* Wr = W + (size_t)row * I;
  const int ka0 = (8*hi) >> 2, ka1 = (16 + 8*hi) >> 2, kb0 = (16*hi) >> 2;
  v8f acc0 = {}, acc1 = {};
  for (int k0 = 0; k0 < I; k0 += 64) {       // I % 64 == 0 for every layer here
    __builtin_prefetch(Wr + k0 + 2048, 0, 0);
    #pragma unroll
    for (int u = 0; u < 2; ++u) {
      const float4* pa = reinterpret_cast<const float4*>(Wr + k0 + 32*u);
      const float4* pb = reinterpret_cast<const float4*>(x + k0 + 32*u);
      v16h a = cvt16(pa[ka0], pa[ka0+1], pa[ka1], pa[ka1+1]);
      v16h b = cvt16(pb[ka0], pb[ka0+1], pb[ka1], pb[ka1+1]);
      if (u == 0) acc0 = __builtin_amdgcn_wmma_f32_16x16x32_f16(false, a, false, b,
                                                                (short)0, acc0, false, false);
      else        acc1 = __builtin_amdgcn_wmma_f32_16x16x32_f16(false, a, false, b,
                                                                (short)0, acc1, false, false);
    }
  }
  v8f acc = acc0 + acc1;
  if (m == 0) {
    #pragma unroll
    for (int r = 0; r < 8; ++r) {
      int orow = blockIdx.x*16 + hi*8 + r;
      if (orow < O) {
        float v = acc[r] + bias[orow];
        if (addv) v += addv[orow];
        if (act == ACT_RELU) v = fmaxf(v, 0.f);
        else if (act == ACT_SIG) v = sigmf(v);
        y[orow] = v * scale + off;
      }
    }
  }
}
#endif

// h_new = (1-z)*n + z*h,  r/z/n from precomputed gx = Wih@x+bih, gh = Whh@h+bhh
__global__ void k_gru_combine(const float* __restrict__ gx, const float* __restrict__ gh,
                              const float* __restrict__ hprev, float* __restrict__ hnew, int H)
{
  int i = blockIdx.x*blockDim.x + threadIdx.x;
  if (i >= H) return;
  float r = sigmf(gx[i] + gh[i]);
  float z = sigmf(gx[H+i] + gh[H+i]);
  float n = tanhf(gx[2*H+i] + r*gh[2*H+i]);
  hnew[i] = (1.f - z)*n + z*hprev[i];
}

// Front-end feature extraction + exp-norm state updates + conv time-buffers out.
__global__ void k_features(const float* __restrict__ spec, const float* __restrict__ mu_in,
                           const float* __restrict__ s_in, float* __restrict__ mu_out,
                           float* __restrict__ s_out, float* __restrict__ fen,
                           float* __restrict__ fse, float* __restrict__ o_erb_buf,
                           float* __restrict__ o_df_buf)
{
  int f = blockIdx.x*blockDim.x + threadIdx.x;
  if (f >= 481) return;
  float sr = spec[f*2], si = spec[f*2+1];
  float mag = sqrtf(sr*sr + si*si + 1e-12f);
  float fe = 10.f * log10f(mag + 1e-10f);
  float mu = 0.99f*mu_in[f] + 0.01f*fe;
  mu_out[f] = mu;
  float v = (fe - mu) / (40.f + 1e-5f);
  fen[f] = v;
  if (f < 480) o_erb_buf[f] = v;               // erb_conv0_buf_out
  if (f < 96) {
    float xab = sqrtf(sr*sr + si*si + 1e-12f);
    float s = 0.99f*s_in[f] + 0.01f*xab;
    s_out[f] = s;
    float den = sqrtf(s + 1e-5f);
    float fr = sr/den, fi = si/den;
    fse[f] = fr; fse[96+f] = fi;               // feat_spec_enc [2][96]
    o_df_buf[f] = fr; o_df_buf[96+f] = fi;     // df_conv0_buf_out
  }
}

// Direct conv over the freq axis; kt in {1,2} via two time rows (x0=t-1, x1=t).
__global__ void k_conv_f(const float* __restrict__ x0, const float* __restrict__ x1,
                         const float* __restrict__ W, const float* __restrict__ bias,
                         float* __restrict__ y, int Ci, int Co, int Fin, int Fout,
                         int kt, int kf, int stride, int fpad, int act)
{
  int idx = blockIdx.x*blockDim.x + threadIdx.x;
  if (idx >= Co*Fout) return;
  int co = idx / Fout, fo = idx % Fout;
  float a = bias[co];
  for (int ci = 0; ci < Ci; ++ci) {
    for (int t = 0; t < kt; ++t) {
      const float* xr = (t == kt-1 ? x1 : x0) + ci*Fin;
      const float* w = W + ((size_t)(co*Ci + ci)*kt + t)*kf;
      for (int j = 0; j < kf; ++j) {
        int fi = fo*stride - fpad + j;
        if (fi >= 0 && fi < Fin) a += w[j]*xr[fi];
      }
    }
  }
  if (act == ACT_RELU) a = fmaxf(a, 0.f);
  else if (act == ACT_SIG) a = sigmf(a);
  y[(size_t)co*Fout + fo] = a;
}

// ConvTranspose over freq, kernel (1,2): y[co][2f+k] = relu(sum_i x[i][f]*W[i][co][k] + b)
__global__ void k_convT_f2(const float* __restrict__ x, const float* __restrict__ W,
                           const float* __restrict__ bias, float* __restrict__ y,
                           int Ci, int Co, int F)
{
  int idx = blockIdx.x*blockDim.x + threadIdx.x;
  if (idx >= Co*2*F) return;
  int co = idx / (2*F), fk = idx % (2*F);
  int f = fk >> 1, k = fk & 1;
  float a = bias[co];
  for (int i = 0; i < Ci; ++i) a += x[i*F + f] * W[((size_t)i*Co + co)*2 + k];
  y[(size_t)co*2*F + fk] = fmaxf(a, 0.f);
}

// Sequential bidirectional GRU scan over freq (one workgroup, weights in LDS).
__global__ void k_gru_scan(const float* __restrict__ x, int F,
                           const float* __restrict__ Wih, const float* __restrict__ Whh,
                           const float* __restrict__ bih, const float* __restrict__ bhh,
                           float* __restrict__ y, int reverse)
{
  __shared__ float sW[96*64], sU[96*32], sbi[96], sbh[96];
  __shared__ float xt[64], h[32], gx[96], gh[96];
  int t = threadIdx.x;
  for (int i = t; i < 96*64; i += blockDim.x) sW[i] = Wih[i];
  for (int i = t; i < 96*32; i += blockDim.x) sU[i] = Whh[i];
  if (t < 96) { sbi[t] = bih[t]; sbh[t] = bhh[t]; }
  if (t < 32) h[t] = 0.f;
  __syncthreads();
  for (int s = 0; s < F; ++s) {
    int f = reverse ? (F-1-s) : s;
    if (t < 64) xt[t] = x[t*F + f];
    __syncthreads();
    if (t < 96) {
      float a = sbi[t]; const float* w = sW + t*64;
      #pragma unroll 8
      for (int j = 0; j < 64; ++j) a += w[j]*xt[j];
      gx[t] = a;
      float b2 = sbh[t]; const float* u = sU + t*32;
      #pragma unroll 8
      for (int j = 0; j < 32; ++j) b2 += u[j]*h[j];
      gh[t] = b2;
    }
    __syncthreads();
    if (t < 32) {
      float r = sigmf(gx[t] + gh[t]);
      float z = sigmf(gx[32+t] + gh[32+t]);
      float n = tanhf(gx[64+t] + r*gh[64+t]);
      float hv = (1.f - z)*n + z*h[t];
      h[t] = hv; y[f*32 + t] = hv;
    }
    __syncthreads();
  }
}

// fc_intra(concat(yf,yb)) -> LayerNorm -> residual add into x (per frequency)
__global__ void k_intra_fuse(const float* __restrict__ yf, const float* __restrict__ yb,
                             const float* __restrict__ Wfc, const float* __restrict__ bfc,
                             const float* __restrict__ g, const float* __restrict__ bln,
                             float* __restrict__ x, int F)
{
  int f = blockIdx.x, c = threadIdx.x;   // 64 threads
  __shared__ float r1[64], r2[64];
  float a = bfc[c];
  const float* w = Wfc + c*64;
  for (int j = 0; j < 32; ++j) a += w[j]    * yf[f*32 + j];
  for (int j = 0; j < 32; ++j) a += w[32+j] * yb[f*32 + j];
  r1[c] = a; r2[c] = a*a;
  __syncthreads();
  for (int s = 32; s > 0; s >>= 1) { if (c < s) { r1[c] += r1[c+s]; r2[c] += r2[c+s]; } __syncthreads(); }
  float mu = r1[0]*(1.f/64.f), var = r2[0]*(1.f/64.f) - mu*mu;
  x[c*F + f] += (a - mu)*rsqrtf(var + 1e-5f)*g[c] + bln[c];
}

// inter GRU cell (per frequency) + fc_inter + LN + residual; emits new hidden state
__global__ void k_inter_fuse(float* __restrict__ x, int F,
                             const float* __restrict__ hprev, float* __restrict__ hnew_out,
                             const float* __restrict__ Wih, const float* __restrict__ Whh,
                             const float* __restrict__ bih, const float* __restrict__ bhh,
                             const float* __restrict__ Wfc, const float* __restrict__ bfc,
                             const float* __restrict__ g, const float* __restrict__ bln)
{
  int f = blockIdx.x, t = threadIdx.x;   // 192 threads
  __shared__ float xe[64], h[64], gx[192], gh[192], hn[64], r1[64], r2[64];
  if (t < 64) { xe[t] = x[t*F + f]; h[t] = hprev[f*64 + t]; }
  __syncthreads();
  {
    float a = bih[t]; const float* w = Wih + t*64;
    for (int j = 0; j < 64; ++j) a += w[j]*xe[j];
    gx[t] = a;
    float b2 = bhh[t]; const float* u = Whh + t*64;
    for (int j = 0; j < 64; ++j) b2 += u[j]*h[j];
    gh[t] = b2;
  }
  __syncthreads();
  if (t < 64) {
    float r = sigmf(gx[t] + gh[t]);
    float z = sigmf(gx[64+t] + gh[64+t]);
    float n = tanhf(gx[128+t] + r*gh[128+t]);
    float hv = (1.f - z)*n + z*h[t];
    hn[t] = hv; hnew_out[f*64 + t] = hv;
  }
  __syncthreads();
  float a = 0.f;
  if (t < 64) {
    a = bfc[t]; const float* w = Wfc + t*64;
    for (int j = 0; j < 64; ++j) a += w[j]*hn[j];
    r1[t] = a; r2[t] = a*a;
  }
  __syncthreads();
  for (int s = 32; s > 0; s >>= 1) { if (t < s) { r1[t] += r1[t+s]; r2[t] += r2[t+s]; } __syncthreads(); }
  if (t < 64) {
    float mu = r1[0]*(1.f/64.f), var = r2[0]*(1.f/64.f) - mu*mu;
    x[t*F + f] += (a - mu)*rsqrtf(var + 1e-5f)*g[t] + bln[t];
  }
}

__global__ void k_pack_fc(const float* __restrict__ x, float* __restrict__ v, int C, int F){
  int i = blockIdx.x*blockDim.x + threadIdx.x;
  if (i >= C*F) return;
  int f = i / C, c = i % C;
  v[i] = x[c*F + f];                                // (C,F) -> flat (F,C)
}
__global__ void k_add_perm(float* __restrict__ t, const float* __restrict__ v, int C, int F){
  int i = blockIdx.x*blockDim.x + threadIdx.x;
  if (i >= C*F) return;
  int c = i / F, f = i % F;
  t[i] += v[f*C + c];                               // t(C,F) += v(F,C)
}
__global__ void k_add(float* __restrict__ y, const float* __restrict__ x, int n){
  int i = blockIdx.x*blockDim.x + threadIdx.x;
  if (i < n) y[i] += x[i];
}
__global__ void k_copy(float* __restrict__ y, const float* __restrict__ x, int n){
  int i = blockIdx.x*blockDim.x + threadIdx.x;
  if (i < n) y[i] = x[i];
}

// mask extend + spec_masked
__global__ void k_mask(const float* __restrict__ spec, const float* __restrict__ m,
                       float* __restrict__ sm){
  int f = blockIdx.x*blockDim.x + threadIdx.x;
  if (f >= 481) return;
  float mm = (f == 480) ? m[478] : m[f];
  sm[f*2]   = spec[f*2]   * mm;
  sm[f*2+1] = spec[f*2+1] * mm;
}

// df_convp (kt=2) + df_out residual + tanh coefs + order-5 complex deep filter.
__global__ void k_df_final(const float* __restrict__ dfo, const float* __restrict__ c0,
                           const float* __restrict__ c0buf, const float* __restrict__ Wp,
                           const float* __restrict__ bp, const float* __restrict__ dfbuf,
                           const float* __restrict__ sm, float* __restrict__ spec_e)
{
  int f = blockIdx.x*blockDim.x + threadIdx.x;
  if (f >= 481) return;
  if (f >= 96) { spec_e[f*2] = sm[f*2]; spec_e[f*2+1] = sm[f*2+1]; return; }
  float co_[10];
  for (int j = 0; j < 10; ++j) {
    float a = bp[j] + dfo[f*10 + j];
    const float* w = Wp + (size_t)j*64*2;
    for (int i = 0; i < 64; ++i)
      a += w[i*2]*c0buf[i*96 + f] + w[i*2+1]*c0[i*96 + f];
    co_[j] = tanhf(a);
  }
  float fr = 0.f, fi = 0.f;
  for (int k = 0; k < 5; ++k) {
    float wr = (k < 4) ? dfbuf[(k*481 + f)*2]     : sm[f*2];
    float wi = (k < 4) ? dfbuf[(k*481 + f)*2 + 1] : sm[f*2+1];
    float cr = co_[2*k], ci = co_[2*k+1];
    fr += wr*cr - wi*ci;
    fi += wr*ci + wi*cr;
  }
  spec_e[f*2] = fr; spec_e[f*2+1] = fi;
}

// ---------------------------------------------------------------------------
// Host orchestration
// ---------------------------------------------------------------------------
static inline dim3 g1(int n){ return dim3((n + 127)/128); }

static void gemv(hipStream_t s, const float* W, const float* x, const float* b,
                 const float* add, float* y, int O, int I, int act = ACT_NONE,
                 float scale = 1.f, float off = 0.f)
{
  k_gemv_wmma<<<dim3((O + 15)/16), dim3(32), 0, s>>>(W, x, b, add, y, O, I, act, scale, off);
}

// Parameter leaf indices assume recursive dict-insertion-order flattening.
#define P_ERB_CONV0 14
#define P_ERB_CONV1 16
#define P_ERB_CONV2 18
#define P_ERB_CONV3 20
#define P_DF_CONV0  22
#define P_DF_CONV1  24
#define P_DPRNN_ERB 26   // in_proj(2) + 2 blocks(20 each) + out_proj(2) = 44 leaves
#define P_DPRNN_DF  70
#define P_ERB_FC_EMB 114
#define P_DF_FC_EMB  116
#define P_EMB_GRU    118 // lin_in W,b | gru Wih,Whh,bih,bhh | lin_out W,b
#define P_LSNR       126
#define P_ERBDEC_GRU 128
#define P_ERBDEC_FC  136
#define P_CONV3P 138
#define P_CONVT3 140
#define P_CONV2P 142
#define P_CONVT2 144
#define P_CONV1P 146
#define P_CONVT1 148
#define P_CONV0P 150
#define P_CONV0_OUT 152
#define P_DF_GRU  154
#define P_DF_SKIP 162
#define P_DF_CONVP 164
#define P_DF_OUT   166

static void run_dprnn(hipStream_t s, const float* const* in, int pb,
                      const float* x, float* xd, float* eout, int F,
                      const float* h0, const float* h1, float* h0o, float* h1o,
                      float* yf, float* yb)
{
  // in_proj (1x1)
  k_conv_f<<<g1(64*F), 128, 0, s>>>(x, x, in[pb], in[pb+1], xd, 64, 64, F, F, 1, 1, 1, 0, ACT_NONE);
  for (int blk = 0; blk < 2; ++blk) {
    int b = pb + 2 + blk*20;
    const float* hp = blk ? h1 : h0;
    float* ho = blk ? h1o : h0o;
    k_gru_scan<<<1, 128, 0, s>>>(xd, F, in[b+0], in[b+1], in[b+2], in[b+3], yf, 0);
    k_gru_scan<<<1, 128, 0, s>>>(xd, F, in[b+4], in[b+5], in[b+6], in[b+7], yb, 1);
    k_intra_fuse<<<F, 64, 0, s>>>(yf, yb, in[b+8], in[b+9], in[b+10], in[b+11], xd, F);
    k_inter_fuse<<<F, 192, 0, s>>>(xd, F, hp, ho, in[b+12], in[b+13], in[b+14], in[b+15],
                                   in[b+16], in[b+17], in[b+18], in[b+19]);
  }
  // out_proj (1x1)
  k_conv_f<<<g1(64*F), 128, 0, s>>>(xd, xd, in[pb+42], in[pb+43], eout, 64, 64, F, F, 1, 1, 1, 0, ACT_NONE);
}

extern "C" void kernel_launch(void* const* d_in, const int* in_sizes, int n_in,
                              void* d_out, int out_size, void* d_ws, size_t ws_size,
                              hipStream_t stream)
{
  (void)in_sizes; (void)n_in; (void)out_size; (void)ws_size;
  const float* const* in = (const float* const*)d_in;
  float* out = (float*)d_out;

  // Output layout (flat concat in reference return order)
  float* o_spec_e   = out;            // 962
  float* o_lsnr     = out + 962;      // 1
  float* o_mu       = out + 963;      // 481
  float* o_s        = out + 1444;     // 96
  float* o_ench     = out + 1540;     // 1024
  float* o_erbdech  = out + 2564;     // 1024
  float* o_dfdech   = out + 3588;     // 1024
  float* o_erbs0    = out + 4612;     // 3840
  float* o_erbs1    = out + 8452;     // 3840
  float* o_dfs0     = out + 12292;    // 3072
  float* o_dfs1     = out + 15364;    // 3072
  float* o_erbc0buf = out + 18436;    // 480
  float* o_dfc0buf  = out + 18916;    // 192
  float* o_dfcpbuf  = out + 19108;    // 6144
  float* o_dfspec   = out + 25252;    // 3848

  // Workspace carving
  float* w = (float*)d_ws;
  size_t off = 0;
  auto alloc = [&](size_t n){ float* p = w + off; off += (n + 7) & ~(size_t)7; return p; };
  float* fen  = alloc(481);
  float* fse  = alloc(192);
  float* e0   = alloc(64*480);
  float* e1   = alloc(64*240);
  float* e2   = alloc(64*120);
  float* e3   = alloc(64*60);
  float* xd_e = alloc(64*60);
  float* e3d  = alloc(64*60);
  float* yf_e = alloc(60*32);
  float* yb_e = alloc(60*32);
  float* c0   = alloc(64*96);
  float* c1   = alloc(64*48);
  float* xd_d = alloc(64*48);
  float* c1d  = alloc(64*48);
  float* yf_d = alloc(48*32);
  float* yb_d = alloc(48*32);
  float* pe3d = alloc(3840);
  float* pc1d = alloc(3072);
  float* cemb = alloc(1024);
  float* emb  = alloc(1024);
  float* xi   = alloc(1024);
  float* gx   = alloc(3072);
  float* gh   = alloc(3072);
  float* emb2 = alloc(1024);
  float* edec = alloc(1024);
  float* v3840= alloc(3840);
  float* skip = alloc(1024);
  float* cdf  = alloc(1024);
  float* dfo  = alloc(960);
  float* t3   = alloc(64*60);
  float* d3   = alloc(64*120);
  float* t2   = alloc(64*120);
  float* d2   = alloc(64*240);
  float* t1   = alloc(64*240);
  float* d1   = alloc(64*480);
  float* t0   = alloc(64*480);
  float* mk   = alloc(480);
  float* sm   = alloc(962);

  // 1. features + normalization state updates + conv0 time-buffers
  k_features<<<g1(481), 128, 0, stream>>>(in[0], in[1], in[2], o_mu, o_s, fen, fse,
                                          o_erbc0buf, o_dfc0buf);
  // 2. ERB encoder convs
  k_conv_f<<<g1(64*480), 128, 0, stream>>>(in[10], fen, in[P_ERB_CONV0], in[P_ERB_CONV0+1],
                                           e0, 1, 64, 480, 480, 2, 3, 1, 1, ACT_RELU);
  k_conv_f<<<g1(64*240), 128, 0, stream>>>(e0, e0, in[P_ERB_CONV1], in[P_ERB_CONV1+1],
                                           e1, 64, 64, 480, 240, 1, 3, 2, 1, ACT_RELU);
  k_conv_f<<<g1(64*120), 128, 0, stream>>>(e1, e1, in[P_ERB_CONV2], in[P_ERB_CONV2+1],
                                           e2, 64, 64, 240, 120, 1, 3, 2, 1, ACT_RELU);
  k_conv_f<<<g1(64*60), 128, 0, stream>>>(e2, e2, in[P_ERB_CONV3], in[P_ERB_CONV3+1],
                                          e3, 64, 64, 120, 60, 1, 3, 2, 1, ACT_RELU);
  // 3. ERB dual-path RNN
  run_dprnn(stream, in, P_DPRNN_ERB, e3, xd_e, e3d, 60, in[6], in[7], o_erbs0, o_erbs1, yf_e, yb_e);
  // 4. DF encoder convs + dual-path RNN
  k_conv_f<<<g1(64*96), 128, 0, stream>>>(in[11], fse, in[P_DF_CONV0], in[P_DF_CONV0+1],
                                          c0, 2, 64, 96, 96, 2, 3, 1, 1, ACT_RELU);
  k_conv_f<<<g1(64*48), 128, 0, stream>>>(c0, c0, in[P_DF_CONV1], in[P_DF_CONV1+1],
                                          c1, 64, 64, 96, 48, 1, 3, 2, 1, ACT_RELU);
  run_dprnn(stream, in, P_DPRNN_DF, c1, xd_d, c1d, 48, in[8], in[9], o_dfs0, o_dfs1, yf_d, yb_d);
  // 5. embedding heads (WMMA GEMVs)
  k_pack_fc<<<g1(3840), 128, 0, stream>>>(e3d, pe3d, 64, 60);
  k_pack_fc<<<g1(3072), 128, 0, stream>>>(c1d, pc1d, 64, 48);
  gemv(stream, in[P_DF_FC_EMB],  pc1d, in[P_DF_FC_EMB+1],  nullptr, cemb, 1024, 3072);
  gemv(stream, in[P_ERB_FC_EMB], pe3d, in[P_ERB_FC_EMB+1], cemb,    emb,  1024, 3840);
  // 6. emb_gru (sgru)
  gemv(stream, in[P_EMB_GRU],   emb,   in[P_EMB_GRU+1], nullptr, xi, 1024, 1024);
  gemv(stream, in[P_EMB_GRU+2], xi,    in[P_EMB_GRU+4], nullptr, gx, 3072, 1024);
  gemv(stream, in[P_EMB_GRU+3], in[3], in[P_EMB_GRU+5], nullptr, gh, 3072, 1024);
  k_gru_combine<<<g1(1024), 128, 0, stream>>>(gx, gh, in[3], o_ench, 1024);
  gemv(stream, in[P_EMB_GRU+6], o_ench, in[P_EMB_GRU+7], nullptr, emb2, 1024, 1024);
  // 7. lsnr head: sigmoid * 50 - 15
  gemv(stream, in[P_LSNR], emb2, in[P_LSNR+1], nullptr, o_lsnr, 1, 1024, ACT_SIG, 50.f, -15.f);
  // 8. erb decoder sgru + fc
  gemv(stream, in[P_ERBDEC_GRU],   emb2,  in[P_ERBDEC_GRU+1], nullptr, xi, 1024, 1024);
  gemv(stream, in[P_ERBDEC_GRU+2], xi,    in[P_ERBDEC_GRU+4], nullptr, gx, 3072, 1024);
  gemv(stream, in[P_ERBDEC_GRU+3], in[4], in[P_ERBDEC_GRU+5], nullptr, gh, 3072, 1024);
  k_gru_combine<<<g1(1024), 128, 0, stream>>>(gx, gh, in[4], o_erbdech, 1024);
  gemv(stream, in[P_ERBDEC_GRU+6], o_erbdech, in[P_ERBDEC_GRU+7], nullptr, edec, 1024, 1024);
  gemv(stream, in[P_ERBDEC_FC], edec, in[P_ERBDEC_FC+1], nullptr, v3840, 3840, 1024);
  // 9. ERB decoder (conv-p skip + convT x2 upsampling chain)
  k_conv_f<<<g1(64*60), 128, 0, stream>>>(e3, e3, in[P_CONV3P], in[P_CONV3P+1],
                                          t3, 64, 64, 60, 60, 1, 1, 1, 0, ACT_RELU);
  k_add_perm<<<g1(64*60), 128, 0, stream>>>(t3, v3840, 64, 60);
  k_convT_f2<<<g1(64*120), 128, 0, stream>>>(t3, in[P_CONVT3], in[P_CONVT3+1], d3, 64, 64, 60);
  k_conv_f<<<g1(64*120), 128, 0, stream>>>(e2, e2, in[P_CONV2P], in[P_CONV2P+1],
                                           t2, 64, 64, 120, 120, 1, 1, 1, 0, ACT_RELU);
  k_add<<<g1(64*120), 128, 0, stream>>>(t2, d3, 64*120);
  k_convT_f2<<<g1(64*240), 128, 0, stream>>>(t2, in[P_CONVT2], in[P_CONVT2+1], d2, 64, 64, 120);
  k_conv_f<<<g1(64*240), 128, 0, stream>>>(e1, e1, in[P_CONV1P], in[P_CONV1P+1],
                                           t1, 64, 64, 240, 240, 1, 1, 1, 0, ACT_RELU);
  k_add<<<g1(64*240), 128, 0, stream>>>(t1, d2, 64*240);
  k_convT_f2<<<g1(64*480), 128, 0, stream>>>(t1, in[P_CONVT1], in[P_CONVT1+1], d1, 64, 64, 240);
  k_conv_f<<<g1(64*480), 128, 0, stream>>>(e0, e0, in[P_CONV0P], in[P_CONV0P+1],
                                           t0, 64, 64, 480, 480, 1, 1, 1, 0, ACT_RELU);
  k_add<<<g1(64*480), 128, 0, stream>>>(t0, d1, 64*480);
  k_conv_f<<<g1(480), 128, 0, stream>>>(t0, t0, in[P_CONV0_OUT], in[P_CONV0_OUT+1],
                                        mk, 64, 1, 480, 480, 1, 3, 1, 1, ACT_SIG);
  k_mask<<<g1(481), 128, 0, stream>>>(in[0], mk, sm);
  // 10. DF decoder sgru + skip + df_out
  gemv(stream, in[P_DF_GRU],   emb2,  in[P_DF_GRU+1], nullptr, xi, 1024, 1024);
  gemv(stream, in[P_DF_GRU+2], xi,    in[P_DF_GRU+4], nullptr, gx, 3072, 1024);
  gemv(stream, in[P_DF_GRU+3], in[5], in[P_DF_GRU+5], nullptr, gh, 3072, 1024);
  k_gru_combine<<<g1(1024), 128, 0, stream>>>(gx, gh, in[5], o_dfdech, 1024);
  gemv(stream, in[P_DF_SKIP], emb2, in[P_DF_SKIP+1], nullptr, skip, 1024, 1024);
  gemv(stream, in[P_DF_GRU+6], o_dfdech, in[P_DF_GRU+7], skip, cdf, 1024, 1024);
  gemv(stream, in[P_DF_OUT], cdf, in[P_DF_OUT+1], nullptr, dfo, 960, 1024);
  // 11. deep-filter application + outputs
  k_df_final<<<g1(481), 128, 0, stream>>>(dfo, c0, in[12], in[P_DF_CONVP], in[P_DF_CONVP+1],
                                          in[13], sm, o_spec_e);
  k_copy<<<g1(6144), 128, 0, stream>>>(o_dfcpbuf, c0, 6144);          // df_convp_buf_out
  k_copy<<<g1(3*962), 128, 0, stream>>>(o_dfspec, in[13] + 962, 3*962); // window rows 1..3
  k_copy<<<g1(962), 128, 0, stream>>>(o_dfspec + 3*962, sm, 962);       // window row 4
}
